// ChannelWiseAttention_39633958208117
// MI455X (gfx1250) — matrix operand order repaired and checked by reading, not compiled, not run on hardware
//
#include <hip/hip_runtime.h>
#include <math.h>

// ---------------- constants -----------------
#define D_MODEL   768
#define NUM_HEADS 6
#define HEAD_DIM  128
#define Bb        8
#define Cc        4096
#define NTOK      (Bb * Cc)          // 32768 tokens
#define NFEAT     (3 * D_MODEL)      // 2304 qkv features
#define KCH       409                // top-k channels
#define M_TILE    32                 // tokens per workgroup
#define N_TILES   (NFEAT / 16)       // 144

// workspace layout (bytes)
#define WBF_BYTES  ((size_t)NFEAT * D_MODEL * 2)         //   3,538,944
#define OUT_BYTES  ((size_t)NTOK * D_MODEL * 4)          // 100,663,296
#define IMP_BYTES  ((size_t)NTOK * 4)                    //     131,072

// LDS carve (bytes)
#define XA_BYTES   (M_TILE * D_MODEL * 2)                // 49152  x tile, bf16
#define QK_BYTES   (M_TILE * 2 * D_MODEL * 2)            // 98304  q+k, bf16
#define VV_BYTES   (M_TILE * D_MODEL * 4)                // 98304  v, fp32
#define GS_BYTES   (M_TILE * NUM_HEADS * 4)              //   768
#define PT_BYTES   (256 * 4)                             //  1024
#define SMEM_TOTAL (XA_BYTES + QK_BYTES + VV_BYTES + GS_BYTES + PT_BYTES) // 247552

typedef __attribute__((ext_vector_type(16))) __bf16 v16bf;
typedef __attribute__((ext_vector_type(8)))  __bf16 v8bf;
typedef __attribute__((ext_vector_type(8)))  float  v8f;

__device__ __forceinline__ unsigned short f2bf_u(float f) {
    unsigned int u = __float_as_uint(f);
    u += 0x7FFFu + ((u >> 16) & 1u);          // round-to-nearest-even
    return (unsigned short)(u >> 16);
}
__device__ __forceinline__ float bf2f(unsigned short h) {
    return __uint_as_float(((unsigned int)h) << 16);
}

// ---------------- kernel 0: W fp32 -> bf16 (stays L2-resident) -------------
__global__ void cwa_convW(const float* __restrict__ W, unsigned short* __restrict__ Wbf) {
    const size_t N = (size_t)NFEAT * D_MODEL;
    for (size_t i = (size_t)blockIdx.x * blockDim.x + threadIdx.x; i < N;
         i += (size_t)gridDim.x * blockDim.x)
        Wbf[i] = f2bf_u(W[i]);
}

// ---------------- kernel 1: fused qkv-GEMM + top-1 head attention ----------
// One workgroup = 32 tokens. 8 waves; each wave owns 18 of the 144 N-tiles and
// runs two 16x16x32 bf16 WMMAs per k-step (M-subtiles 0..15 and 16..31 share
// the B fragment => half the L2 traffic for W).
__global__ void __launch_bounds__(256)
cwa_fused(const float* __restrict__ x, const unsigned short* __restrict__ Wbf,
          const float* __restrict__ bias, float* __restrict__ out_ws,
          float* __restrict__ imp) {
    extern __shared__ __align__(16) char smem[];
    unsigned short* xA  = (unsigned short*)smem;                       // [32][768] bf16
    unsigned short* qk  = (unsigned short*)(smem + XA_BYTES);          // [32][1536] bf16
    float*          vls = (float*)(smem + XA_BYTES + QK_BYTES);        // [32][768] f32
    int*            gst = (int*)(smem + XA_BYTES + QK_BYTES + VV_BYTES);
    float*          prt = (float*)(smem + XA_BYTES + QK_BYTES + VV_BYTES + GS_BYTES);

    const int tid  = threadIdx.x;
    const int wave = tid >> 5;
    const int lane = tid & 31;
    const int n    = lane & 15;      // N column / A row within half
    const int half = lane >> 4;      // lane group select (ISA 7.12.2 layouts)
    const int n0   = blockIdx.x * M_TILE;   // first token of tile

    // ---- phase 1: stage x tile as bf16 (rows are contiguous in global) ----
    for (int i = tid; i < M_TILE * D_MODEL; i += 256)
        xA[i] = f2bf_u(x[(size_t)n0 * D_MODEL + i]);
    __syncthreads();

    // ---- phase 2: qkv = x @ W^T + b via WMMA bf16 ----
    for (int nt = wave; nt < N_TILES; nt += 8) {
        v8f acc0 = {};
        v8f acc1 = {};
        const unsigned short* wrow = Wbf + (size_t)(nt * 16 + n) * D_MODEL;
        if (nt + 8 < N_TILES)   // next W row this wave will touch -> global_prefetch
            __builtin_prefetch(Wbf + (size_t)((nt + 8) * 16 + n) * D_MODEL, 0, 3);

        for (int k0 = 0; k0 < D_MODEL; k0 += 32) {
            // B fragment: 16 bf16 of one W row per lane, table k order (matches A)
            const v8bf blo = *(const v8bf*)(wrow + k0 + half * 8);
            const v8bf bhi = *(const v8bf*)(wrow + k0 + 16 + half * 8);
            const v16bf Bf = __builtin_shufflevector(blo, bhi,
                0,1,2,3,4,5,6,7,8,9,10,11,12,13,14,15);
            // A fragment, M-subtile 0 (tokens 0..15)
            const v8bf a0l = *(const v8bf*)(xA + n * D_MODEL + k0 + half * 8);
            const v8bf a0h = *(const v8bf*)(xA + n * D_MODEL + k0 + 16 + half * 8);
            const v16bf A0 = __builtin_shufflevector(a0l, a0h,
                0,1,2,3,4,5,6,7,8,9,10,11,12,13,14,15);
            // A fragment, M-subtile 1 (tokens 16..31)
            const v8bf a1l = *(const v8bf*)(xA + (16 + n) * D_MODEL + k0 + half * 8);
            const v8bf a1h = *(const v8bf*)(xA + (16 + n) * D_MODEL + k0 + 16 + half * 8);
            const v16bf A1 = __builtin_shufflevector(a1l, a1h,
                0,1,2,3,4,5,6,7,8,9,10,11,12,13,14,15);

            acc0 = __builtin_amdgcn_wmma_f32_16x16x32_bf16(
                false, A0, false, Bf, (short)0, acc0, false, false);
            acc1 = __builtin_amdgcn_wmma_f32_16x16x32_bf16(
                false, A1, false, Bf, (short)0, acc1, false, false);
        }

        const int col = nt * 16 + n;          // qkv feature index
        const float bv = bias[col];
        if (nt < 96) {                        // q or k features -> bf16 LDS
            #pragma unroll
            for (int r = 0; r < 8; ++r) {
                const int m = r + 8 * half;
                qk[m * (2 * D_MODEL) + col]        = f2bf_u(acc0[r] + bv);
                qk[(m + 16) * (2 * D_MODEL) + col] = f2bf_u(acc1[r] + bv);
            }
        } else {                              // v features -> fp32 LDS
            const int vc = col - 2 * D_MODEL;
            #pragma unroll
            for (int r = 0; r < 8; ++r) {
                const int m = r + 8 * half;
                vls[m * D_MODEL + vc]        = acc0[r] + bv;
                vls[(m + 16) * D_MODEL + vc] = acc1[r] + bv;
            }
        }
    }
    __syncthreads();

    // ---- phase 3: scores + top-1 head (softmax of 1 element == 1.0) ----
    if (tid < M_TILE * NUM_HEADS) {
        const int t = tid / NUM_HEADS, h = tid % NUM_HEADS;
        const unsigned short* qp = qk + t * (2 * D_MODEL) + h * HEAD_DIM;
        int best = 0; float bvs = -INFINITY;
        for (int g = 0; g < NUM_HEADS; ++g) {
            const unsigned short* kp = qk + t * (2 * D_MODEL) + D_MODEL + g * HEAD_DIM;
            float s = 0.f;
            for (int d = 0; d < HEAD_DIM; ++d)
                s = fmaf(bf2f(qp[d]), bf2f(kp[d]), s);
            if (s > bvs) { bvs = s; best = g; }   // ties -> lowest g (lax.top_k)
        }
        gst[tid] = best;
    }
    __syncthreads();

    // ---- phase 4: out[n, d*6+h] = v[n, g*, d]; channel importance ----
    {
        const int t = tid >> 3, sub = tid & 7;     // 8 threads / token
        float ss = 0.f;
        for (int c = sub; c < D_MODEL; c += 8) {
            const int d = c / NUM_HEADS, h = c % NUM_HEADS;
            const float v = vls[t * D_MODEL + gst[t * NUM_HEADS + h] * HEAD_DIM + d];
            out_ws[(size_t)(n0 + t) * D_MODEL + c] = v;
            ss = fmaf(v, v, ss);
        }
        prt[tid] = ss;
    }
    __syncthreads();
    if (tid < M_TILE) {
        float s = 0.f;
        for (int i = 0; i < 8; ++i) s += prt[tid * 8 + i];
        imp[n0 + tid] = sqrtf(s);
    }
}

// ---------------- kernel 2: per-batch top-409 channels ---------------------
__global__ void cwa_topk(const float* __restrict__ imp, int* __restrict__ chidx) {
    __shared__ float vals[Cc];
    __shared__ float rv[256];
    __shared__ int   ri[256];
    const int b = blockIdx.x, tid = threadIdx.x;
    for (int i = tid; i < Cc; i += 256) vals[i] = imp[b * Cc + i];
    __syncthreads();
    for (int j = 0; j < KCH; ++j) {
        float bv = -INFINITY; int bi = Cc;
        for (int i = tid; i < Cc; i += 256) {
            const float v = vals[i];
            if (v > bv) { bv = v; bi = i; }       // strided scan ascending keeps min idx
        }
        rv[tid] = bv; ri[tid] = bi;
        __syncthreads();
        for (int s = 128; s > 0; s >>= 1) {
            if (tid < s) {
                const bool take = (rv[tid + s] > rv[tid]) ||
                                  (rv[tid + s] == rv[tid] && ri[tid + s] < ri[tid]);
                if (take) { rv[tid] = rv[tid + s]; ri[tid] = ri[tid + s]; }
            }
            __syncthreads();
        }
        if (tid == 0) { chidx[b * KCH + j] = ri[0]; vals[ri[0]] = -INFINITY; }
        __syncthreads();
    }
}

// ---------------- kernel 3: gather sparse rows + emit indices --------------
__global__ void cwa_gather(const float* __restrict__ out_ws,
                           const int* __restrict__ chidx,
                           float* __restrict__ dout, int out_size) {
    const int j = blockIdx.x, b = blockIdx.y, tid = threadIdx.x;
    const int c = chidx[b * KCH + j];
    const float* src = out_ws + ((size_t)b * Cc + c) * D_MODEL;
    const size_t dbase = ((size_t)b * KCH + j) * D_MODEL;
    for (int k = tid; k < D_MODEL; k += 256)
        if (dbase + k < (size_t)out_size) dout[dbase + k] = src[k];
    const int F = Bb * KCH * D_MODEL;             // 2,512,896
    if (tid == 0) {
        const int p = F + b * KCH + j;
        if (p < out_size) dout[p] = (float)c;     // ch_idx
        if (b == 0 && j == 0 && F + Bb * KCH < out_size)
            dout[F + Bb * KCH] = (float)KCH;      // scalar k_channels
    }
}

// ---------------- launcher -------------------------------------------------
extern "C" void kernel_launch(void* const* d_in, const int* in_sizes, int n_in,
                              void* d_out, int out_size, void* d_ws, size_t ws_size,
                              hipStream_t stream) {
    (void)in_sizes; (void)n_in; (void)ws_size;
    const float* x = (const float*)d_in[0];
    const float* W = (const float*)d_in[1];
    const float* b = (const float*)d_in[2];

    char* ws = (char*)d_ws;
    unsigned short* Wbf    = (unsigned short*)ws;
    float*          out_ws = (float*)(ws + WBF_BYTES);
    float*          imp    = (float*)(ws + WBF_BYTES + OUT_BYTES);
    int*            chidx  = (int*)(ws + WBF_BYTES + OUT_BYTES + IMP_BYTES);

    cwa_convW<<<1728, 256, 0, stream>>>(W, Wbf);
    cwa_fused<<<NTOK / M_TILE, 256, SMEM_TOTAL, stream>>>(x, Wbf, b, out_ws, imp);
    cwa_topk<<<Bb, 256, 0, stream>>>(imp, chidx);
    dim3 gg(KCH, Bb);
    cwa_gather<<<gg, 256, 0, stream>>>(out_ws, chidx, (float*)d_out, out_size);
}
